// CausalSelfAttention_70411693850717
// MI455X (gfx1250) — matrix operand order repaired and checked by reading, not compiled, not run on hardware
//
#include <hip/hip_runtime.h>
#include <hip/hip_bf16.h>

// ---- problem constants -----------------------------------------------------
constexpr int Bn  = 4;
constexpr int Tn  = 2048;
constexpr int Cn  = 256;
constexpr int Hn  = 8;
constexpr int HDn = 32;           // == K-dim of bf16 WMMA -> perfect fit
constexpr int Mrows = Bn * Tn;    // 8192
constexpr float SCALE = 0.17677669529663687f;  // 1/sqrt(32)

typedef __bf16 bf16;
typedef __attribute__((ext_vector_type(16))) __bf16 v16bf;  // 32B aligned
typedef __attribute__((ext_vector_type(8)))  __bf16 v8bf;   // 16B aligned
typedef __attribute__((ext_vector_type(8)))  float  v8f;
typedef __attribute__((ext_vector_type(4)))  unsigned int v4u;
typedef __attribute__((ext_vector_type(8)))  int v8i_;
typedef __attribute__((ext_vector_type(4)))  int v4i_;

// ---- WMMA fragment helpers (contiguous k order: k = 16*(lane/16) + s) ------
__device__ inline v16bf load_a_frag(const bf16* base, int ld, int m0, int kt) {
  const int lane = threadIdx.x & 31;
  const int row  = lane & 15;
  const int kb   = (lane < 16) ? 0 : 16;
  const bf16* p  = base + (size_t)(m0 + row) * ld + kt + kb;
  v8bf lo = *(const v8bf*)(p);        // K = kb..kb+7
  v8bf hi = *(const v8bf*)(p + 8);    // K = kb+8..kb+15
  return __builtin_shufflevector(lo, hi, 0, 1, 2, 3, 4, 5, 6, 7,
                                          8, 9, 10, 11, 12, 13, 14, 15);
}

// B (32x16): lane -> N = lane%16, K = (lane<16?0:16)+j. N-major source gives
// one contiguous 32B-aligned 16-elem chunk per lane.
__device__ inline v16bf load_b_contig(const bf16* base, size_t row_stride,
                                      int n0, int kt) {
  const int lane = threadIdx.x & 31;
  const bf16* p = base + (size_t)(n0 + (lane & 15)) * row_stride + kt +
                  ((lane < 16) ? 0 : 16);
  return *(const v16bf*)p;
}

__device__ inline v8f wmma_bf16(v16bf a, v16bf b, v8f c) {
  return __builtin_amdgcn_wmma_f32_16x16x32_bf16(false, a, false, b, (short)0,
                                                 c, false, false);
}

__device__ inline float rmax16(float v) {
#pragma unroll
  for (int m = 1; m <= 8; m <<= 1) v = fmaxf(v, __shfl_xor(v, m, 32));
  return v;
}
__device__ inline float rsum16(float v) {
#pragma unroll
  for (int m = 1; m <= 8; m <<= 1) v += __shfl_xor(v, m, 32);
  return v;
}

// ---- TDM: tensor_load_to_lds with hand-built D# (CDNA5 ISA) ----
// 1-D copy: tile_d1 = 0 (dim1 unused).  2-D: stride0 = elements between rows.
__device__ inline void tdm_load(const bf16* gaddr, unsigned lds_byte_off,
                                unsigned tile_d0, unsigned tile_d1,
                                unsigned tensor_d0, unsigned tensor_d1,
                                unsigned stride0) {
  const unsigned long long ga = (unsigned long long)(uintptr_t)gaddr;
  v4u g0;
  g0[0] = 1u;                                    // count=1, user descriptor
  g0[1] = lds_byte_off;                          // lds_addr (bytes)
  g0[2] = (unsigned)(ga & 0xffffffffu);          // global_addr[31:0]
  g0[3] = (unsigned)((ga >> 32) & 0x01ffffffu)   // global_addr[56:32]
          | (2u << 30);                          // type = 2 ("image")
  v8i_ g1;
  g1[0] = (int)(1u << 16);                       // data_size=1 (2 bytes)
  g1[1] = (int)((tensor_d0 & 0xffffu) << 16);    // tensor_dim0[15:0]
  g1[2] = (int)((tensor_d0 >> 16) | ((tensor_d1 & 0xffffu) << 16));
  g1[3] = (int)((tensor_d1 >> 16) | (tile_d0 << 16));   // tile_dim0
  g1[4] = (int)(tile_d1 & 0xffffu);              // tile_dim1 (0 => 1-D)
  g1[5] = (int)stride0;                          // tensor_dim0_stride[31:0]
  g1[6] = 0;
  g1[7] = 0;
  v4i_ z4 = {0, 0, 0, 0};
#if defined(__clang_major__) && __clang_major__ >= 23
  v8i_ z8 = {0, 0, 0, 0, 0, 0, 0, 0};
  __builtin_amdgcn_tensor_load_to_lds(g0, g1, z4, z4, z8, 0);
#else
  __builtin_amdgcn_tensor_load_to_lds(g0, g1, z4, z4, 0);
#endif
}

// ---- kernel 0a: f32 -> bf16 convert ----------------------------------------
__global__ void convert_bf16_kernel(const float* __restrict__ in,
                                    bf16* __restrict__ out, int n) {
  int i = blockIdx.x * blockDim.x + threadIdx.x;
  int stride = gridDim.x * blockDim.x;
  for (; i < n; i += stride) out[i] = (bf16)in[i];
}

// ---- kernel 0b: f32 [256][N] -> bf16 [N][256] transpose --------------------
__global__ void convert_transpose_kernel(const float* __restrict__ in,
                                         bf16* __restrict__ out, int N) {
  int idx = blockIdx.x * blockDim.x + threadIdx.x;
  const int total = Cn * N;
  const int stride = gridDim.x * blockDim.x;
  for (; idx < total; idx += stride) {
    const int n = idx >> 8, k = idx & (Cn - 1);
    out[idx] = (bf16)in[(size_t)k * N + n];
  }
}

// ---- kernel 1: QKV GEMM + bias; Q,K -> [B,H,T,HD], V -> [B,H,HD,T] bf16 ----
__global__ __launch_bounds__(256) void qkv_kernel(
    const bf16* __restrict__ xb, const bf16* __restrict__ wT,  // wT: [768][256]
    const float* __restrict__ bias,
    bf16* __restrict__ Qb, bf16* __restrict__ Kb, bf16* __restrict__ Vt) {
  const int wave = threadIdx.x >> 5, lane = threadIdx.x & 31;
  const int gw = blockIdx.x * 8 + wave;
  const int mt = gw / 12, ng = gw % 12;
  const int m0 = mt * 16, n0 = ng * 64;

  v8f acc[4] = {v8f{}, v8f{}, v8f{}, v8f{}};
#pragma unroll
  for (int kt = 0; kt < Cn; kt += 32) {
    v16bf a = load_a_frag(xb, Cn, m0, kt);
#pragma unroll
    for (int t = 0; t < 4; ++t) {
      v16bf b = load_b_contig(wT, Cn, n0 + 16 * t, kt);
      acc[t] = wmma_bf16(a, b, acc[t]);
    }
  }

  const int rbase = (lane < 16) ? 0 : 8;
  const int ncol  = lane & 15;
  const int b_    = m0 >> 11;
  const int trow0 = (m0 & (Tn - 1)) + rbase;
  const int which = n0 >> 8;                   // 0=Q, 1=K, 2=V (wave-uniform)

  if (which == 2) {
#pragma unroll
    for (int t = 0; t < 4; ++t) {
      const int c = (n0 + 16 * t) & (Cn - 1);
      const int h = c >> 5, d = (c & (HDn - 1)) + ncol;
      const float bv = bias[n0 + 16 * t + ncol];
      v8bf pk;
#pragma unroll
      for (int r = 0; r < 8; ++r) pk[r] = (bf16)(acc[t][r] + bv);
      *(v8bf*)(Vt + ((size_t)(b_ * Hn + h) * HDn + d) * Tn + trow0) = pk;
    }
  } else {
    bf16* dst = (which == 0) ? Qb : Kb;
#pragma unroll
    for (int t = 0; t < 4; ++t) {
      const int c = (n0 + 16 * t) & (Cn - 1);
      const int h = c >> 5, d = (c & (HDn - 1)) + ncol;
      const float bv = bias[n0 + 16 * t + ncol];
      const size_t base = ((size_t)(b_ * Hn + h) * Tn + trow0) * HDn + d;
#pragma unroll
      for (int r = 0; r < 8; ++r)
        dst[base + (size_t)r * HDn] = (bf16)(acc[t][r] + bv);
    }
  }
}

// ---- attention 64-key block: operands come from LDS (TDM-staged) -----------
template <bool MASKED>
__device__ inline void attn_block(const v16bf& aq, const bf16* kb,
                                  const bf16* vb, bf16* pw,
                                  int j0, int q0, int rbase, int ncol,
                                  v8f& o0, v8f& o1, float* mrow, float* lrow) {
  v8f s[4];
#pragma unroll
  for (int u = 0; u < 4; ++u)
    s[u] = wmma_bf16(aq, load_b_contig(kb, HDn, 16 * u, 0), v8f{});

#pragma unroll
  for (int r = 0; r < 8; ++r) {
    float a0 = s[0][r] * SCALE, a1 = s[1][r] * SCALE;
    float a2 = s[2][r] * SCALE, a3 = s[3][r] * SCALE;
    if (MASKED) {
      const int q = q0 + rbase + r;
      const float NI = -__builtin_inff();
      if (j0 + ncol      > q) a0 = NI;
      if (j0 + 16 + ncol > q) a1 = NI;
      if (j0 + 32 + ncol > q) a2 = NI;
      if (j0 + 48 + ncol > q) a3 = NI;
    }
    const float tmax = rmax16(fmaxf(fmaxf(a0, a1), fmaxf(a2, a3)));
    const float newm = fmaxf(mrow[r], tmax);
    const float corr = __expf(mrow[r] - newm);
    const float p0 = __expf(a0 - newm), p1 = __expf(a1 - newm);
    const float p2 = __expf(a2 - newm), p3 = __expf(a3 - newm);
    lrow[r] = lrow[r] * corr + rsum16((p0 + p1) + (p2 + p3));
    mrow[r] = newm;
    o0[r] *= corr;  o1[r] *= corr;
    const int ro = (rbase + r) * 64;
    pw[ro + ncol]      = (bf16)p0;
    pw[ro + 16 + ncol] = (bf16)p1;
    pw[ro + 32 + ncol] = (bf16)p2;
    pw[ro + 48 + ncol] = (bf16)p3;
  }

  const v16bf ap0 = load_a_frag(pw, 64, 0, 0);
  const v16bf ap1 = load_a_frag(pw, 64, 0, 32);
  o0 = wmma_bf16(ap0, load_b_contig(vb, 64, 0,  0),  o0);
  o1 = wmma_bf16(ap0, load_b_contig(vb, 64, 16, 0),  o1);
  o0 = wmma_bf16(ap1, load_b_contig(vb, 64, 0,  32), o0);
  o1 = wmma_bf16(ap1, load_b_contig(vb, 64, 16, 32), o1);
}

// ---- kernel 2: flash attention; block = one (b,h) x 128 queries ------------
// Wave 0 TDM-DMAs the shared 64-key K/V tiles into double-buffered LDS while
// all 8 waves consume the previous tile; sync = s_wait_tensorcnt + barriers.
__global__ __launch_bounds__(256) void attn_kernel(
    const bf16* __restrict__ Qb, const bf16* __restrict__ Kb,
    const bf16* __restrict__ Vt, bf16* __restrict__ Yb) {
  __shared__ __align__(16) bf16 kbuf[2][64 * HDn];   // [key][dim]   4 KB x2
  __shared__ __align__(16) bf16 vbuf[2][HDn * 64];   // [dim][key]   4 KB x2
  __shared__ __align__(16) bf16 lds_p[8][16 * 64];   // per-wave P tile

  const int wave = threadIdx.x >> 5, lane = threadIdx.x & 31;
  const int bh  = blockIdx.x >> 4;                   // 0..31, shared by block
  const int qtb = (blockIdx.x & 15) * 8;             // first q-tile of block
  const int qtile = qtb + wave;
  const bf16* Qh = Qb + (size_t)bh * Tn * HDn;       // [T][HD]
  const bf16* Kh = Kb + (size_t)bh * Tn * HDn;       // [T][HD]
  const bf16* Vh = Vt + (size_t)bh * HDn * Tn;       // [HD][T]
  const int q0 = qtile * 16;
  const int ntiles = (qtb * 16 + 127) / 64 + 1;      // block-level tile count

  const v16bf aq = load_a_frag(Qh, HDn, q0, 0);

  v8f o0 = v8f{}, o1 = v8f{};
  float mrow[8], lrow[8];
#pragma unroll
  for (int r = 0; r < 8; ++r) { mrow[r] = -__builtin_inff(); lrow[r] = 0.0f; }
  const int rbase = (lane < 16) ? 0 : 8;
  const int ncol  = lane & 15;
  bf16* pw = lds_p[wave];

  // preload tile 0
  if (wave == 0) {
    tdm_load(Kh, (unsigned)(uintptr_t)(void*)kbuf[0],
             64 * HDn, 0, 64 * HDn, 1, 64 * HDn);          // 1-D, contiguous
    tdm_load(Vh, (unsigned)(uintptr_t)(void*)vbuf[0],
             64, HDn, Tn, HDn, Tn);                        // 2-D, stride T
    __builtin_amdgcn_s_wait_tensorcnt(0);
  }
  __syncthreads();

  int cur = 0;
  for (int i = 0; i < ntiles; ++i) {
    const int j0 = i * 64;
    if (wave == 0 && i + 1 < ntiles) {                     // DMA next tile
      const int jn = j0 + 64;
      tdm_load(Kh + (size_t)jn * HDn, (unsigned)(uintptr_t)(void*)kbuf[cur ^ 1],
               64 * HDn, 0, 64 * HDn, 1, 64 * HDn);
      tdm_load(Vh + jn, (unsigned)(uintptr_t)(void*)vbuf[cur ^ 1],
               64, HDn, Tn, HDn, Tn);
    }
    if (j0 <= q0 + 15) {                                   // wave-uniform
      if (j0 + 63 <= q0)
        attn_block<false>(aq, kbuf[cur], vbuf[cur], pw, j0, q0, rbase, ncol,
                          o0, o1, mrow, lrow);
      else
        attn_block<true>(aq, kbuf[cur], vbuf[cur], pw, j0, q0, rbase, ncol,
                         o0, o1, mrow, lrow);
    }
    __syncthreads();                                       // done reading cur
    if (wave == 0 && i + 1 < ntiles) __builtin_amdgcn_s_wait_tensorcnt(0);
    __syncthreads();                                       // next tile visible
    cur ^= 1;
  }

  // --- finalize & write Y [B*T][C] bf16 ---
  const int b_ = bh / Hn, h = bh % Hn;
#pragma unroll
  for (int r = 0; r < 8; ++r) {
    const float inv = 1.0f / lrow[r];
    const size_t row = (size_t)(b_ * Tn + q0 + rbase + r) * Cn + h * HDn;
    Yb[row + ncol]      = (bf16)(o0[r] * inv);
    Yb[row + 16 + ncol] = (bf16)(o1[r] * inv);
  }
}

// ---- kernel 3: output projection, f32 result -------------------------------
__global__ __launch_bounds__(256) void proj_kernel(
    const bf16* __restrict__ Yb, const bf16* __restrict__ wpT,  // [256][256]
    const float* __restrict__ bias, float* __restrict__ out) {
  const int wave = threadIdx.x >> 5, lane = threadIdx.x & 31;
  const int gw = blockIdx.x * 8 + wave;
  const int mt = gw >> 2, ng = gw & 3;
  const int m0 = mt * 16, n0 = ng * 64;

  v8f acc[4] = {v8f{}, v8f{}, v8f{}, v8f{}};
#pragma unroll
  for (int kt = 0; kt < Cn; kt += 32) {
    v16bf a = load_a_frag(Yb, Cn, m0, kt);
#pragma unroll
    for (int t = 0; t < 4; ++t) {
      v16bf b = load_b_contig(wpT, Cn, n0 + 16 * t, kt);
      acc[t] = wmma_bf16(a, b, acc[t]);
    }
  }
  const int rbase = (lane < 16) ? 0 : 8;
  const int ncol  = lane & 15;
#pragma unroll
  for (int t = 0; t < 4; ++t) {
    const int n = n0 + 16 * t + ncol;
    const float bv = bias[n];
#pragma unroll
    for (int r = 0; r < 8; ++r)
      out[(size_t)(m0 + rbase + r) * Cn + n] = acc[t][r] + bv;
  }
}

// ---- host launcher ---------------------------------------------------------
extern "C" void kernel_launch(void* const* d_in, const int* in_sizes, int n_in,
                              void* d_out, int out_size, void* d_ws, size_t ws_size,
                              hipStream_t stream) {
  const float* x      = (const float*)d_in[0];
  const float* w_qkv  = (const float*)d_in[1];
  const float* b_qkv  = (const float*)d_in[2];
  const float* w_proj = (const float*)d_in[3];
  const float* b_proj = (const float*)d_in[4];
  float* out = (float*)d_out;

  size_t off = 0;
  auto carve = [&](size_t bytes) {
    void* p = (char*)d_ws + off;
    off += (bytes + 255) & ~(size_t)255;
    return p;
  };
  bf16* xb     = (bf16*)carve((size_t)Mrows * Cn * 2);
  bf16* wqkvT  = (bf16*)carve((size_t)3 * Cn * Cn * 2);
  bf16* wprojT = (bf16*)carve((size_t)Cn * Cn * 2);
  bf16* Qb     = (bf16*)carve((size_t)Mrows * Cn * 2);
  bf16* Kb     = (bf16*)carve((size_t)Mrows * Cn * 2);
  bf16* Vt     = (bf16*)carve((size_t)Mrows * Cn * 2);
  bf16* Yb     = (bf16*)carve((size_t)Mrows * Cn * 2);

  convert_bf16_kernel<<<1024, 256, 0, stream>>>(x, xb, Mrows * Cn);
  convert_transpose_kernel<<<192, 256, 0, stream>>>(w_qkv, wqkvT, 3 * Cn);
  convert_transpose_kernel<<<64, 256, 0, stream>>>(w_proj, wprojT, Cn);

  qkv_kernel<<<768, 256, 0, stream>>>(xb, wqkvT, b_qkv, Qb, Kb, Vt);
  attn_kernel<<<512, 256, 0, stream>>>(Qb, Kb, Vt, Yb);
  proj_kernel<<<256, 256, 0, stream>>>(Yb, wprojT, b_proj, out);
}
